// DecoderBlock_37254546326206
// MI455X (gfx1250) — matrix operand order repaired and checked by reading, not compiled, not run on hardware
//
#include <hip/hip_runtime.h>

typedef __bf16 bf16;
typedef __bf16 v16bf __attribute__((ext_vector_type(16)));
typedef float  v8f   __attribute__((ext_vector_type(8)));
typedef unsigned u32x4 __attribute__((ext_vector_type(4)));
typedef int      i32x4 __attribute__((ext_vector_type(4)));
typedef int      i32x8 __attribute__((ext_vector_type(8)));

#define D_MODEL 768
#define D_FF    3072
#define N_HEADS 12
#define DK      64
#define SEQ     2048
#define BATCH   2
#define ROWS    (BATCH*SEQ)   // 4096
#define ASTRIDE 40            // As row stride (elems) = 20 dwords = 16 data + 4 pad

#if defined(__has_builtin)
#if __has_builtin(__builtin_amdgcn_tensor_load_to_lds)
#define HAVE_TDM 1
#endif
#endif
#ifndef HAVE_TDM
#define HAVE_TDM 0
#endif

// ---------------------------------------------------------------- fp32->bf16
__global__ void cvt_f32_bf16(const float* __restrict__ in, bf16* __restrict__ out, int n) {
    int i = blockIdx.x * blockDim.x + threadIdx.x;
    if (i < n) out[i] = (bf16)in[i];
}

// ---------------------------------------------------------------- WMMA GEMM
// C[M,N] = A[M,K](bf16) * B[K,N](bf16) + bias (+resid) (+relu)
// block tile 128x64, 8 waves as 4x2, wave tile 32x32 (4 WMMA per K-step of 32).
// A tile staged by the Tensor Data Mover (TDM descriptor, hardware LDS padding
// to ASTRIDE), falling back to global_load_async_to_lds_b128 if TDM builtin is
// absent. B tile staged through VGPRs with a register transpose.
template<int RELU, int RESID, int OUTBF>
__global__ __launch_bounds__(256)
void gemm_wmma(const bf16* __restrict__ A, const bf16* __restrict__ B,
               const float* __restrict__ bias, const float* __restrict__ resid,
               float* __restrict__ outF, bf16* __restrict__ outB,
               int M, int N, int K)
{
    __shared__ bf16 As[128][ASTRIDE];   // [m][k]
    __shared__ bf16 Bs[64][36];         // transposed: [n][k]

    const int m0 = blockIdx.y * 128;
    const int n0 = blockIdx.x * 64;
    const int t    = threadIdx.x;
    const int lane = t & 31, wid = t >> 5;
    const int wm = (wid >> 1) * 32;   // wave row offset inside tile
    const int wn = (wid & 1) * 32;    // wave col offset inside tile
    const int lm = lane & 15, kh = lane >> 4;

    v8f c00 = {}, c01 = {}, c10 = {}, c11 = {};
    const unsigned ldsA = (unsigned)(size_t)&As[0][0];

    for (int kb = 0; kb < K; kb += 32) {
#if HAVE_TDM
        // ---- A tile 128x32 via Tensor Data Mover (one DMA per K-step)
        if (wid == 0) {
            size_t ga = (size_t)(A + (size_t)m0 * K + kb);   // tile start (bytes)
            u32x4 g0;
            g0[0] = 1u;                                       // count=1, user mode
            g0[1] = ldsA;                                     // lds_addr
            g0[2] = (unsigned)(ga & 0xffffffffu);             // global_addr[31:0]
            g0[3] = (unsigned)((ga >> 32) & 0x1ffffffu)       // global_addr[56:32]
                  | (2u << 30);                               // type=2 ("image")
            i32x8 g1;
            g1[0] = (int)((1u << 16)      // data_size = 2 bytes
                  | (1u << 20)            // pad_enable
                  | (3u << 22)            // pad_interval = 16 dwords (one tile row)
                  | (3u << 25));          // pad_amount   = 4 dwords (8 elems)
            g1[1] = (int)(((unsigned)K & 0xffffu) << 16);                       // tensor_dim0 lo
            g1[2] = (int)((((unsigned)K >> 16) & 0xffffu)
                  | (((unsigned)M & 0xffffu) << 16));                           // dim0 hi | dim1 lo
            g1[3] = (int)((((unsigned)M >> 16) & 0xffffu) | (32u << 16));       // dim1 hi | tile_dim0=32
            g1[4] = 128;                                                        // tile_dim1=128, tile_dim2=0
            g1[5] = K;                                                          // tensor_dim0_stride lo
            g1[6] = 0;
            g1[7] = 0;
            i32x4 z4 = {0, 0, 0, 0};
#if __clang_major__ >= 23
            i32x8 z8 = {0, 0, 0, 0, 0, 0, 0, 0};
            __builtin_amdgcn_tensor_load_to_lds(g0, g1, z4, z4, z8, 0);
#else
            __builtin_amdgcn_tensor_load_to_lds(g0, g1, z4, z4, 0);
#endif
            __builtin_amdgcn_s_wait_tensorcnt(0);
        }
#else
        // ---- fallback: async DMA to LDS, 2 x b128 chunks per thread
        #pragma unroll
        for (int i = 0; i < 2; ++i) {
            int ch  = t + 256 * i;
            int row = ch >> 2, col = (ch & 3) * 8;
            const bf16* gp = A + (size_t)(m0 + row) * K + kb + col;
            unsigned lp = (unsigned)(size_t)&As[row][col];
            asm volatile("global_load_async_to_lds_b128 %0, %1, off"
                         :: "v"(lp), "v"(gp) : "memory");
        }
        asm volatile("s_wait_asynccnt 0x0" ::: "memory");
#endif
        // ---- B tile 32x64, register-transposed into Bs[n][k]
        {
            int row = t >> 3, col = (t & 7) * 8;
            union { uint4 u; bf16 h[8]; } v;
            v.u = *(const uint4*)(B + (size_t)(kb + row) * N + n0 + col);
            #pragma unroll
            for (int e = 0; e < 8; ++e) Bs[col + e][row] = v.h[e];
        }
        // ---- prefetch next K-tile into cache while this one is consumed
        if (kb + 32 < K) {
            int row = t >> 2, col = (t & 3) * 8;
            __builtin_prefetch(A + (size_t)(m0 + row) * K + kb + 32 + col, 0, 1);
            int rb = t >> 3, cb = (t & 7) * 8;
            __builtin_prefetch(B + (size_t)(kb + 32 + rb) * N + n0 + cb, 0, 1);
        }
        __syncthreads();

        // A fragments (16x32 bf16): lane holds row M=lm, K-halves selected by kh
        v16bf a0, a1;
        #pragma unroll
        for (int v = 0; v < 8; ++v) {
            int kk = (v < 4 ? 2 * v : 16 + 2 * (v - 4)) + 8 * kh;
            a0[2 * v]     = As[wm + lm][kk];
            a0[2 * v + 1] = As[wm + lm][kk + 1];
            a1[2 * v]     = As[wm + 16 + lm][kk];
            a1[2 * v + 1] = As[wm + 16 + lm][kk + 1];
        }
        // B fragments (32x16): lane holds col N, K=8*kh..8*kh+7 and 16+8*kh..16+8*kh+7
        v16bf b0v, b1v;
        #pragma unroll
        for (int e = 0; e < 16; ++e) {
            int kk = 8 * kh + e + (e < 8 ? 0 : 8);
            b0v[e] = Bs[wn + lm][kk];
            b1v[e] = Bs[wn + 16 + lm][kk];
        }
        c00 = __builtin_amdgcn_wmma_f32_16x16x32_bf16(false, a0, false, b0v, (short)0, c00, false, false);
        c01 = __builtin_amdgcn_wmma_f32_16x16x32_bf16(false, a0, false, b1v, (short)0, c01, false, false);
        c10 = __builtin_amdgcn_wmma_f32_16x16x32_bf16(false, a1, false, b0v, (short)0, c10, false, false);
        c11 = __builtin_amdgcn_wmma_f32_16x16x32_bf16(false, a1, false, b1v, (short)0, c11, false, false);
        __syncthreads();
    }

    // Epilogue. C layout: VGPR r -> M = r + 8*kh, N = lane&15
    #pragma unroll
    for (int ij = 0; ij < 4; ++ij) {
        const v8f& c = ij == 0 ? c00 : ij == 1 ? c01 : ij == 2 ? c10 : c11;
        int i = ij >> 1, j = ij & 1;
        int nn = n0 + wn + j * 16 + lm;
        float bv = bias[nn];
        #pragma unroll
        for (int r = 0; r < 8; ++r) {
            int mm = m0 + wm + i * 16 + r + 8 * kh;
            float val = c[r] + bv;
            if (RESID) val += resid[(size_t)mm * N + nn];
            if (RELU)  val = val > 0.f ? val : 0.f;
            if (OUTBF) outB[(size_t)mm * N + nn] = (bf16)val;
            else       outF[(size_t)mm * N + nn] = val;
        }
    }
}

// ---------------------------------------------------------------- attention
// One wave per (batch, head, 16-query tile). Flash-style online softmax.
// Q/K/V tiles staged with async LDS DMA (ASYNCcnt).
__global__ __launch_bounds__(32)
void attn_kernel(const bf16* __restrict__ Q, const bf16* __restrict__ Km,
                 const bf16* __restrict__ Vm, bf16* __restrict__ O)
{
    __shared__ bf16 Qs[16][64];   // 128B rows: async chunks stay 16B aligned
    __shared__ bf16 Ks[32][72];   // 144B rows
    __shared__ bf16 Vs[32][72];
    __shared__ bf16 Ps[16][34];

    const int lane = threadIdx.x;
    const int qt = blockIdx.x, h = blockIdx.y, b = blockIdx.z;
    const int m0 = qt * 16;
    const int lm = lane & 15, kh = lane >> 4;
    const size_t base = ((size_t)b * SEQ) * D_MODEL + (size_t)h * DK;

    // stage Q tile 16x64 (4 async b128 chunks per lane)
    #pragma unroll
    for (int i = 0; i < 4; ++i) {
        int cidx = lane + 32 * i;
        int row = cidx >> 3, col = (cidx & 7) * 8;
        const bf16* gp = Q + base + (size_t)(m0 + row) * D_MODEL + col;
        unsigned lp = (unsigned)(size_t)&Qs[row][col];
        asm volatile("global_load_async_to_lds_b128 %0, %1, off"
                     :: "v"(lp), "v"(gp) : "memory");
    }
    asm volatile("s_wait_asynccnt 0x0" ::: "memory");
    __syncthreads();

    v16bf qa[2];
    #pragma unroll
    for (int f = 0; f < 2; ++f)
        #pragma unroll
        for (int v = 0; v < 8; ++v) {
            int kk = f * 32 + (v < 4 ? 2 * v : 16 + 2 * (v - 4)) + 8 * kh;
            qa[f][2 * v]     = Qs[lm][kk];
            qa[f][2 * v + 1] = Qs[lm][kk + 1];
        }

    v8f o0 = {}, o1 = {}, o2 = {}, o3 = {};
    float mi[8], li[8];
    #pragma unroll
    for (int r = 0; r < 8; ++r) { mi[r] = -3.0e38f; li[r] = 0.f; }

    const int ktEnd = (m0 + 16 + 31) >> 5;   // causal: only key blocks <= diagonal
    for (int kt = 0; kt < ktEnd; ++kt) {
        int n0 = kt * 32;
        __syncthreads();
        // stage K and V blocks 32x64 via async LDS DMA (8 chunks each per lane)
        #pragma unroll
        for (int i = 0; i < 8; ++i) {
            int cidx = lane + 32 * i;
            int row = cidx >> 3, col = (cidx & 7) * 8;
            const bf16* gk = Km + base + (size_t)(n0 + row) * D_MODEL + col;
            const bf16* gv = Vm + base + (size_t)(n0 + row) * D_MODEL + col;
            unsigned lk = (unsigned)(size_t)&Ks[row][col];
            unsigned lv = (unsigned)(size_t)&Vs[row][col];
            asm volatile("global_load_async_to_lds_b128 %0, %1, off"
                         :: "v"(lk), "v"(gk) : "memory");
            asm volatile("global_load_async_to_lds_b128 %0, %1, off"
                         :: "v"(lv), "v"(gv) : "memory");
        }
        if (kt + 1 < ktEnd) {   // prefetch next key/value block
            __builtin_prefetch(Km + base + (size_t)(n0 + 32 + lane) * D_MODEL, 0, 1);
            __builtin_prefetch(Vm + base + (size_t)(n0 + 32 + lane) * D_MODEL, 0, 1);
        }
        asm volatile("s_wait_asynccnt 0x0" ::: "memory");
        __syncthreads();

        // scores S = Q * K^T : B-matrix[feat][key] == Ks[key][feat]
        v8f s0 = {}, s1 = {};
        #pragma unroll
        for (int fk = 0; fk < 2; ++fk) {
            v16bf bk0, bk1;
            #pragma unroll
            for (int e = 0; e < 16; ++e) {
                int kk = fk * 32 + 8 * kh + e + (e < 8 ? 0 : 8);
                bk0[e] = Ks[lm][kk];
                bk1[e] = Ks[16 + lm][kk];
            }
            s0 = __builtin_amdgcn_wmma_f32_16x16x32_bf16(false, qa[fk], false, bk0, (short)0, s0, false, false);
            s1 = __builtin_amdgcn_wmma_f32_16x16x32_bf16(false, qa[fk], false, bk1, (short)0, s1, false, false);
        }

        // causal mask + scale + online softmax (rows live across 16-lane groups)
        float fac[8], p0[8], p1[8];
        #pragma unroll
        for (int r = 0; r < 8; ++r) {
            int row = m0 + r + 8 * kh;
            float a0 = ((n0 + lm)      <= row) ? s0[r] * 0.125f : -3.0e38f;
            float a1 = ((n0 + 16 + lm) <= row) ? s1[r] * 0.125f : -3.0e38f;
            float mx = fmaxf(a0, a1);
            mx = fmaxf(mx, __shfl_xor(mx, 1, 16));
            mx = fmaxf(mx, __shfl_xor(mx, 2, 16));
            mx = fmaxf(mx, __shfl_xor(mx, 4, 16));
            mx = fmaxf(mx, __shfl_xor(mx, 8, 16));
            float mnew = fmaxf(mi[r], mx);
            float f  = __expf(mi[r] - mnew);
            float e0 = __expf(a0 - mnew);
            float e1 = __expf(a1 - mnew);
            float sum = e0 + e1;
            sum += __shfl_xor(sum, 1, 16);
            sum += __shfl_xor(sum, 2, 16);
            sum += __shfl_xor(sum, 4, 16);
            sum += __shfl_xor(sum, 8, 16);
            li[r] = li[r] * f + sum;
            mi[r] = mnew;
            fac[r] = f;
            p0[r] = e0; p1[r] = e1;
        }

        // relayout P (C-frag) -> A-frag via LDS; rescale O accumulators
        #pragma unroll
        for (int r = 0; r < 8; ++r) {
            Ps[r + 8 * kh][lm]      = (bf16)p0[r];
            Ps[r + 8 * kh][16 + lm] = (bf16)p1[r];
            o0[r] *= fac[r]; o1[r] *= fac[r]; o2[r] *= fac[r]; o3[r] *= fac[r];
        }
        __syncthreads();

        v16bf pa;
        #pragma unroll
        for (int v = 0; v < 8; ++v) {
            int kk = (v < 4 ? 2 * v : 16 + 2 * (v - 4)) + 8 * kh;
            pa[2 * v]     = Ps[lm][kk];
            pa[2 * v + 1] = Ps[lm][kk + 1];
        }
        // O += P(16x32) * V(32x64): B-matrix[key][feat] == Vs row-major (column gather)
        #pragma unroll
        for (int fn = 0; fn < 4; ++fn) {
            v16bf vb;
            #pragma unroll
            for (int e = 0; e < 16; ++e)
                vb[e] = Vs[8 * kh + e + (e < 8 ? 0 : 8)][fn * 16 + lm];
            v8f& oo = (fn == 0 ? o0 : fn == 1 ? o1 : fn == 2 ? o2 : o3);
            oo = __builtin_amdgcn_wmma_f32_16x16x32_bf16(false, pa, false, vb, (short)0, oo, false, false);
        }
    }

    // normalize and store bf16
    #pragma unroll
    for (int fn = 0; fn < 4; ++fn) {
        const v8f& oo = (fn == 0 ? o0 : fn == 1 ? o1 : fn == 2 ? o2 : o3);
        #pragma unroll
        for (int r = 0; r < 8; ++r) {
            int row = m0 + r + 8 * kh;
            int col = fn * 16 + lm;
            O[base + (size_t)row * D_MODEL + col] = (bf16)(oo[r] / li[r]);
        }
    }
}

// ---------------------------------------------------------------- layernorm
__global__ __launch_bounds__(256)
void layernorm_kernel(const float* __restrict__ in, const float* __restrict__ g,
                      const float* __restrict__ be, float* __restrict__ outF,
                      bf16* __restrict__ outB)
{
    __shared__ float red[256];
    const int row = blockIdx.x;
    const int t = threadIdx.x;
    const float* x = in + (size_t)row * D_MODEL;

    float s = 0.f;
    for (int i = t; i < D_MODEL; i += 256) s += x[i];
    red[t] = s; __syncthreads();
    for (int off = 128; off > 0; off >>= 1) { if (t < off) red[t] += red[t + off]; __syncthreads(); }
    float mu = red[0] / D_MODEL;
    __syncthreads();

    float v = 0.f;
    for (int i = t; i < D_MODEL; i += 256) { float d = x[i] - mu; v += d * d; }
    red[t] = v; __syncthreads();
    for (int off = 128; off > 0; off >>= 1) { if (t < off) red[t] += red[t + off]; __syncthreads(); }
    float rs = rsqrtf(red[0] / D_MODEL + 1e-5f);

    for (int i = t; i < D_MODEL; i += 256) {
        float y = (x[i] - mu) * rs * g[i] + be[i];
        outF[(size_t)row * D_MODEL + i] = y;
        if (outB) outB[(size_t)row * D_MODEL + i] = (bf16)y;
    }
}

// ---------------------------------------------------------------- launch
extern "C" void kernel_launch(void* const* d_in, const int* in_sizes, int n_in,
                              void* d_out, int out_size, void* d_ws, size_t ws_size,
                              hipStream_t stream)
{
    const float* x   = (const float*)d_in[0];
    const float* wq  = (const float*)d_in[1];
    const float* bq  = (const float*)d_in[2];
    const float* wk  = (const float*)d_in[3];
    const float* bk  = (const float*)d_in[4];
    const float* wv  = (const float*)d_in[5];
    const float* bv  = (const float*)d_in[6];
    const float* wo  = (const float*)d_in[7];
    const float* bo  = (const float*)d_in[8];
    const float* w1  = (const float*)d_in[9];
    const float* b1  = (const float*)d_in[10];
    const float* w2  = (const float*)d_in[11];
    const float* b2  = (const float*)d_in[12];
    const float* g1  = (const float*)d_in[13];
    const float* be1 = (const float*)d_in[14];
    const float* g2  = (const float*)d_in[15];
    const float* be2 = (const float*)d_in[16];

    char* ws = (char*)d_ws;
    size_t off = 0;
    auto alloc = [&](size_t bytes) -> void* {
        void* p = ws + off;
        off += (bytes + 255) & ~(size_t)255;
        return p;
    };
    bf16*  xb  = (bf16*)alloc((size_t)ROWS * D_MODEL * 2);
    bf16*  wqb = (bf16*)alloc((size_t)D_MODEL * D_MODEL * 2);
    bf16*  wkb = (bf16*)alloc((size_t)D_MODEL * D_MODEL * 2);
    bf16*  wvb = (bf16*)alloc((size_t)D_MODEL * D_MODEL * 2);
    bf16*  wob = (bf16*)alloc((size_t)D_MODEL * D_MODEL * 2);
    bf16*  w1b = (bf16*)alloc((size_t)D_MODEL * D_FF * 2);
    bf16*  w2b = (bf16*)alloc((size_t)D_FF * D_MODEL * 2);
    bf16*  qb  = (bf16*)alloc((size_t)ROWS * D_MODEL * 2);
    bf16*  kb2 = (bf16*)alloc((size_t)ROWS * D_MODEL * 2);
    bf16*  vb2 = (bf16*)alloc((size_t)ROWS * D_MODEL * 2);
    bf16*  ab  = (bf16*)alloc((size_t)ROWS * D_MODEL * 2);
    float* z1  = (float*)alloc((size_t)ROWS * D_MODEL * 4);
    float* h1  = (float*)alloc((size_t)ROWS * D_MODEL * 4);
    bf16*  h1b = (bf16*)alloc((size_t)ROWS * D_MODEL * 2);
    bf16*  f1  = (bf16*)alloc((size_t)ROWS * D_FF * 2);
    float* z2  = (float*)alloc((size_t)ROWS * D_MODEL * 4);

    auto cvt = [&](const float* src, bf16* dst, int n) {
        cvt_f32_bf16<<<(n + 255) / 256, 256, 0, stream>>>(src, dst, n);
    };
    cvt(x,  xb,  ROWS * D_MODEL);
    cvt(wq, wqb, D_MODEL * D_MODEL);
    cvt(wk, wkb, D_MODEL * D_MODEL);
    cvt(wv, wvb, D_MODEL * D_MODEL);
    cvt(wo, wob, D_MODEL * D_MODEL);
    cvt(w1, w1b, D_MODEL * D_FF);
    cvt(w2, w2b, D_FF * D_MODEL);

    dim3 gqkv(D_MODEL / 64, ROWS / 128);
    gemm_wmma<0,0,1><<<gqkv, 256, 0, stream>>>(xb, wqb, bq, nullptr, nullptr, qb,  ROWS, D_MODEL, D_MODEL);
    gemm_wmma<0,0,1><<<gqkv, 256, 0, stream>>>(xb, wkb, bk, nullptr, nullptr, kb2, ROWS, D_MODEL, D_MODEL);
    gemm_wmma<0,0,1><<<gqkv, 256, 0, stream>>>(xb, wvb, bv, nullptr, nullptr, vb2, ROWS, D_MODEL, D_MODEL);

    attn_kernel<<<dim3(SEQ / 16, N_HEADS, BATCH), 32, 0, stream>>>(qb, kb2, vb2, ab);

    gemm_wmma<0,1,0><<<gqkv, 256, 0, stream>>>(ab, wob, bo, x, z1, nullptr, ROWS, D_MODEL, D_MODEL);
    layernorm_kernel<<<ROWS, 256, 0, stream>>>(z1, g1, be1, h1, h1b);

    gemm_wmma<1,0,1><<<dim3(D_FF / 64, ROWS / 128), 256, 0, stream>>>(h1b, w1b, b1, nullptr, nullptr, f1, ROWS, D_FF, D_MODEL);
    gemm_wmma<0,1,0><<<gqkv, 256, 0, stream>>>(f1, w2b, b2, h1, z2, nullptr, ROWS, D_MODEL, D_FF);

    layernorm_kernel<<<ROWS, 256, 0, stream>>>(z2, g2, be2, (float*)d_out, nullptr);
}
